// GraphAttentionLayer_37211596652860
// MI455X (gfx1250) — matrix operand order repaired and checked by reading, not compiled, not run on hardware
//
#include <hip/hip_runtime.h>

typedef __bf16 bf16;
typedef __bf16 bf16x8  __attribute__((ext_vector_type(8)));
typedef __bf16 bf16x16 __attribute__((ext_vector_type(16)));
typedef float  f32x4   __attribute__((ext_vector_type(4)));
typedef float  f32x8   __attribute__((ext_vector_type(8)));
typedef int    i32x4   __attribute__((ext_vector_type(4)));

#define BATCH 16
#define NNODE 2048
#define DIM   256
#define WT_STRIDE 264   // padded bf16 stride for W^T in LDS (avoids bank conflicts)
#define WH_PAD    40    // padded bf16 stride for Wh chunk in LDS
#define LEAKY 0.2f

#if __has_builtin(__builtin_amdgcn_global_load_async_to_lds_b128)
#define HAVE_ASYNC_LDS 1
#else
#define HAVE_ASYNC_LDS 0
#endif

__device__ __forceinline__ void wait_asynccnt(int n)
{
#if __has_builtin(__builtin_amdgcn_s_wait_asynccnt)
    if (n == 0) __builtin_amdgcn_s_wait_asynccnt(0);
    else        __builtin_amdgcn_s_wait_asynccnt(4);
#else
    if (n == 0) asm volatile("s_wait_asynccnt 0x0" ::: "memory");
    else        asm volatile("s_wait_asynccnt 0x4" ::: "memory");
#endif
}

#if HAVE_ASYNC_LDS
// Builtin signature (from toolchain diagnostic): (v4i AS1* src, v4i AS3* dst, imm, imm)
typedef __attribute__((address_space(1))) i32x4* gvec_ptr;
typedef __attribute__((address_space(3))) i32x4* lvec_ptr;

__device__ __forceinline__ void async_cp16(const bf16* g, bf16* l)
{
    // GLOBAL_LOAD_ASYNC_TO_LDS_B128: memory -> LDS, tracked by ASYNCcnt
    const i32x4* gv = (const i32x4*)g;
    i32x4*       lv = (i32x4*)l;
    __builtin_amdgcn_global_load_async_to_lds_b128((gvec_ptr)gv, (lvec_ptr)lv, 0, 0);
}
__device__ __forceinline__ void stage_async(bf16* ldsbase, const bf16* gsrc)
{
    async_cp16(gsrc,      ldsbase);
    async_cp16(gsrc + 8,  ldsbase + 8);
    async_cp16(gsrc + 16, ldsbase + 16);
    async_cp16(gsrc + 24, ldsbase + 24);
}
#endif

// ---------------------------------------------------------------------------
// Kernel 1: Wh = h @ W (bf16 WMMA, fp32 accum), write WhT[b][d][j] in bf16,
//           and f1 = Wh@a1, f2 = Wh@a2 (fp32).
// Grid: (N/128, B), block 256 (8 waves x 16 rows each).
// ---------------------------------------------------------------------------
__global__ void __launch_bounds__(256)
gat_wh_f(const float* __restrict__ h, const float* __restrict__ W,
         const float* __restrict__ a, bf16* __restrict__ WhT,
         float* __restrict__ f1, float* __restrict__ f2)
{
    extern __shared__ bf16 ldsWT[];           // [DIM][WT_STRIDE], W transposed

    const int tid  = threadIdx.x;
    const int b    = blockIdx.y;
    const int i0   = blockIdx.x * 128;
    const int wave = tid >> 5;
    const int lane = tid & 31;
    const int m    = lane & 15;               // matrix column (B/C/D) or A row
    const int hi   = lane >> 4;               // lane half
    const int khalf = hi * 8;                 // A-frag K sub-offset

    // Stage W^T into LDS as bf16: ldsWT[n][k] = W[k][n]
    for (int idx = tid; idx < DIM * DIM; idx += 256) {
        int k = idx >> 8, n = idx & 255;      // coalesced read of W[k][n]
        ldsWT[n * WT_STRIDE + k] = (bf16)W[idx];
    }
    __syncthreads();

    // A fragments: h rows (i0 + wave*16 + m), all K=256 as 8 chunks of 32.
    // ISA 16-bit A layout: lane m=row; e<8 -> K=khalf+e, e>=8 -> K=16+khalf+(e-8)
    const float* hrow = h + ((size_t)(b * NNODE + i0 + wave * 16 + m)) * DIM;
    bf16x16 afrag[8];
    #pragma unroll
    for (int kc = 0; kc < 8; ++kc) {
        f32x4 l0 = *(const f32x4*)(hrow + kc * 32 + khalf);
        f32x4 l1 = *(const f32x4*)(hrow + kc * 32 + khalf + 4);
        f32x4 h0 = *(const f32x4*)(hrow + kc * 32 + 16 + khalf);
        f32x4 h1 = *(const f32x4*)(hrow + kc * 32 + 16 + khalf + 4);
        #pragma unroll
        for (int e = 0; e < 4; ++e) {
            afrag[kc][e]      = (bf16)l0[e];
            afrag[kc][e + 4]  = (bf16)l1[e];
            afrag[kc][e + 8]  = (bf16)h0[e];
            afrag[kc][e + 12] = (bf16)h1[e];
        }
    }

    float f1acc[8] = {0, 0, 0, 0, 0, 0, 0, 0};
    float f2acc[8] = {0, 0, 0, 0, 0, 0, 0, 0};

    #pragma unroll 1
    for (int t = 0; t < 16; ++t) {            // 16 output column tiles
        f32x8 acc = {0.f, 0.f, 0.f, 0.f, 0.f, 0.f, 0.f, 0.f};
        #pragma unroll
        for (int kc = 0; kc < 8; ++kc) {
            // B frag (32x16): lane col n=m, K(e) = kc*32 + (e<8?0:16) + hi*8 + (e&7)
            const bf16* bp = &ldsWT[(t * 16 + m) * WT_STRIDE + kc * 32 + hi * 8];
            bf16x8 blo = *(const bf16x8*)bp;
            bf16x8 bhi = *(const bf16x8*)(bp + 16);
            bf16x16 bfrag;
            #pragma unroll
            for (int e = 0; e < 8; ++e) { bfrag[e] = blo[e]; bfrag[e + 8] = bhi[e]; }
            acc = __builtin_amdgcn_wmma_f32_16x16x32_bf16(
                      false, afrag[kc], false, bfrag, (short)0, acc, false, false);
        }
        // f partials: lane holds column n = t*16+m for rows r+8*hi
        const float a1v = a[t * 16 + m];
        const float a2v = a[DIM + t * 16 + m];
        bf16x8 st;
        #pragma unroll
        for (int r = 0; r < 8; ++r) {
            f1acc[r] += acc[r] * a1v;
            f2acc[r] += acc[r] * a2v;
            st[r] = (bf16)acc[r];
        }
        // Store transposed: WhT[b][n][node], 8 consecutive nodes -> one b128 store
        *(bf16x8*)(WhT + ((size_t)(b * DIM + t * 16 + m)) * NNODE
                   + i0 + wave * 16 + 8 * hi) = st;
    }

    // Reduce f partials across the 16 lanes sharing each row (same hi half)
    #pragma unroll
    for (int r = 0; r < 8; ++r) {
        float v1 = f1acc[r], v2 = f2acc[r];
        #pragma unroll
        for (int sh = 1; sh < 16; sh <<= 1) {
            v1 += __shfl_xor(v1, sh, 32);
            v2 += __shfl_xor(v2, sh, 32);
        }
        f1acc[r] = v1; f2acc[r] = v2;
    }
    if (m == 0) {
        float* o1 = f1 + b * NNODE + i0 + wave * 16 + 8 * hi;
        float* o2 = f2 + b * NNODE + i0 + wave * 16 + 8 * hi;
        #pragma unroll
        for (int r = 0; r < 8; ++r) { o1[r] = f1acc[r]; o2[r] = f2acc[r]; }
    }
}

// ---------------------------------------------------------------------------
// Kernel 2: fused masked-softmax attention: out = elu(softmax(e) @ Wh).
// One pass over adj (the 268 MB stream), P tiles built in registers,
// P @ Wh via bf16 WMMA. Wh slabs are double-buffered in LDS and filled with
// GLOBAL_LOAD_ASYNC_TO_LDS_B128 one chunk ahead (ASYNCcnt-tracked), so the
// copy latency hides under the 16 WMMAs + exps of the current chunk.
// ---------------------------------------------------------------------------
__device__ __forceinline__ float edge_p(int av, float fi, float fj, float& psum)
{
    float s = fi + fj;
    s = (s > 0.f) ? s : LEAKY * s;            // leaky_relu
    float p = (av > 0) ? __expf(s) : 0.f;     // mask -> exp(-1e9) == 0
    psum += p;
    return p;
}

__global__ void __launch_bounds__(256)
gat_attn(const int* __restrict__ adj, const bf16* __restrict__ WhT,
         const float* __restrict__ f1, const float* __restrict__ f2,
         float* __restrict__ out)
{
    __shared__ float f2buf[NNODE];            // 8 KB
    __shared__ bf16  whbuf[2][DIM * WH_PAD];  // 2 x 20 KB double buffer
    __shared__ float sbuf[8][16];

    const int tid  = threadIdx.x;
    const int b    = blockIdx.y;
    const int i0   = blockIdx.x * 128;
    const int wave = tid >> 5;
    const int lane = tid & 31;
    const int m    = lane & 15;
    const int hi   = lane >> 4;
    const int khalf = hi * 8;

    for (int j = tid; j < NNODE; j += 256) f2buf[j] = f2[b * NNODE + j];

    const int   row    = i0 + wave * 16 + m;
    const float fi     = f1[b * NNODE + row];
    const int*  adjrow = adj + ((size_t)(b * NNODE + row)) * NNODE;
    const bf16* whsrc  = WhT + ((size_t)(b * DIM + tid)) * NNODE;  // one d-row/thread
    bf16* mylds0 = &whbuf[0][tid * WH_PAD];
    bf16* mylds1 = &whbuf[1][tid * WH_PAD];

    f32x8 acc[16] = {};                       // 16x256 fp32 accumulators
    float psum = 0.f;

#if HAVE_ASYNC_LDS
    stage_async(mylds0, whsrc);               // prologue: chunk 0 -> buf 0
#endif
    int cur = 0;

    for (int jc = 0; jc < NNODE; jc += 32) {
        __syncthreads();                      // all waves done reading buf[1-cur]
#if HAVE_ASYNC_LDS
        const bool more = (jc + 32 < NNODE);
        if (more) stage_async(cur ? mylds0 : mylds1, whsrc + jc + 32);
        wait_asynccnt(more ? 4 : 0);          // oldest 4 (current chunk) complete
#else
        {   // synchronous fallback: stage current chunk into buf[cur]
            bf16* dst = cur ? mylds1 : mylds0;
            bf16x8 w0 = *(const bf16x8*)(whsrc + jc);
            bf16x8 w1 = *(const bf16x8*)(whsrc + jc + 8);
            bf16x8 w2 = *(const bf16x8*)(whsrc + jc + 16);
            bf16x8 w3 = *(const bf16x8*)(whsrc + jc + 24);
            *(bf16x8*)(dst)      = w0;
            *(bf16x8*)(dst + 8)  = w1;
            *(bf16x8*)(dst + 16) = w2;
            *(bf16x8*)(dst + 24) = w3;
        }
#endif
        if (jc + 32 < NNODE)
            __builtin_prefetch(adjrow + jc + 32, 0, 0);   // global_prefetch_b8
        __syncthreads();                      // buf[cur] visible to all waves

        // Build P A-fragment (16x32): lane m=row, K(e) = (e<8?0:16)+khalf+(e&7)
        i32x4 a0 = *(const i32x4*)(adjrow + jc + khalf);
        i32x4 a1 = *(const i32x4*)(adjrow + jc + khalf + 4);
        i32x4 a2 = *(const i32x4*)(adjrow + jc + 16 + khalf);
        i32x4 a3 = *(const i32x4*)(adjrow + jc + 16 + khalf + 4);
        f32x4 g0 = *(const f32x4*)(&f2buf[jc + khalf]);
        f32x4 g1 = *(const f32x4*)(&f2buf[jc + khalf + 4]);
        f32x4 g2 = *(const f32x4*)(&f2buf[jc + 16 + khalf]);
        f32x4 g3 = *(const f32x4*)(&f2buf[jc + 16 + khalf + 4]);

        bf16x16 pfrag;
        #pragma unroll
        for (int e = 0; e < 4; ++e) {
            pfrag[e]      = (bf16)edge_p(a0[e], fi, g0[e], psum);
            pfrag[e + 4]  = (bf16)edge_p(a1[e], fi, g1[e], psum);
            pfrag[e + 8]  = (bf16)edge_p(a2[e], fi, g2[e], psum);
            pfrag[e + 12] = (bf16)edge_p(a3[e], fi, g3[e], psum);
        }

        // acc[t] += P(16x32) @ Wh(32x16) for each of 16 d-tiles
        const bf16* base = cur ? &whbuf[1][0] : &whbuf[0][0];
        #pragma unroll
        for (int t = 0; t < 16; ++t) {
            const bf16* bp = &base[(t * 16 + m) * WH_PAD + hi * 8];
            bf16x8 blo = *(const bf16x8*)bp;
            bf16x8 bh2 = *(const bf16x8*)(bp + 16);
            bf16x16 bfrag;
            #pragma unroll
            for (int e = 0; e < 8; ++e) { bfrag[e] = blo[e]; bfrag[e + 8] = bh2[e]; }
            acc[t] = __builtin_amdgcn_wmma_f32_16x16x32_bf16(
                         false, pfrag, false, bfrag, (short)0, acc[t], false, false);
        }
        cur ^= 1;
    }

    // Row sums: lanes m and m+16 hold complementary j-subsets of row m
    psum += __shfl_xor(psum, 16, 32);
    if (hi == 0) sbuf[wave][m] = psum;
    __syncthreads();
    float srow[8];
    #pragma unroll
    for (int r = 0; r < 8; ++r) srow[r] = sbuf[wave][8 * hi + r];

    // Normalize + ELU + store (C/D layout: row = r + 8*hi, col = t*16 + m)
    #pragma unroll
    for (int t = 0; t < 16; ++t) {
        #pragma unroll
        for (int r = 0; r < 8; ++r) {
            float sden = srow[r];
            float o = (sden > 0.f) ? acc[t][r] / sden : 0.f;
            o = (o > 0.f) ? o : (__expf(o) - 1.f);
            out[((size_t)(b * NNODE + i0 + wave * 16 + 8 * hi + r)) * DIM
                + t * 16 + m] = o;
        }
    }
}

// ---------------------------------------------------------------------------
extern "C" void kernel_launch(void* const* d_in, const int* in_sizes, int n_in,
                              void* d_out, int out_size, void* d_ws, size_t ws_size,
                              hipStream_t stream)
{
    (void)in_sizes; (void)n_in; (void)out_size; (void)ws_size;
    const float* h   = (const float*)d_in[0];
    const int*   adj = (const int*)d_in[1];
    const float* W   = (const float*)d_in[2];
    const float* a   = (const float*)d_in[3];
    float*       out = (float*)d_out;

    // Workspace: WhT bf16 [B][DIM][NNODE] (16.8 MB) + f1,f2 fp32 [B][NNODE]
    bf16*  WhT = (bf16*)d_ws;
    float* f1  = (float*)((char*)d_ws + (size_t)BATCH * DIM * NNODE * sizeof(bf16));
    float* f2  = f1 + BATCH * NNODE;

    dim3 grid(NNODE / 128, BATCH);
    size_t lds1 = (size_t)DIM * WT_STRIDE * sizeof(bf16);   // 135168 B < 320 KB
    gat_wh_f<<<grid, 256, lds1, stream>>>(h, W, a, WhT, f1, f2);
    gat_attn<<<grid, 256, 0, stream>>>(adj, WhT, f1, f2, out);
}
